// Llama4TextMoe_37151467110424
// MI455X (gfx1250) — matrix operand order repaired and checked by reading, not compiled, not run on hardware
//
#include <hip/hip_runtime.h>
#include <hip/hip_bf16.h>

typedef _Float16 f16;
typedef __attribute__((ext_vector_type(16))) _Float16 v16h;
typedef __attribute__((ext_vector_type(8)))  float    v8f;

#define Tt 4096
#define Hh 2048
#define Ii 1024
#define Ee 16

#define BM 128
#define BN 64
#define BK 32
#define LDK 36   // padded K-stride (halves): 72B rows -> 8B-aligned chunks, bank spread

// ---------------- WMMA helpers (CDNA5 v_wmma_f32_16x16x32_f16) ----------------
__device__ inline v8f wmma32(v16h a, v16h b, v8f c) {
    return __builtin_amdgcn_wmma_f32_16x16x32_f16(
        /*neg_a=*/false, a, /*neg_b=*/false, b,
        /*c_mod=*/(short)0, c, /*reuse_a=*/false, /*reuse_b=*/false);
}

// A fragment (16x32 f16). ISA layout: lanes 0-15: M=lane, K = {0..7, 16..23};
// lanes 16-31: M=lane-16, K = {8..15, 24..31}. tile points at row 0 of the
// 16-row subtile in LDS ([M][LDK] layout, K contiguous).
__device__ inline v16h load_a_frag(const f16* tile, int lane) {
    const f16* p = tile + (lane & 15) * LDK;
    const int hs = (lane >> 4) << 3;
    v16h a;
#pragma unroll
    for (int i = 0; i < 8; i++) a[i] = p[hs + i];
#pragma unroll
    for (int i = 0; i < 8; i++) a[8 + i] = p[16 + hs + i];
    return a;
}

// B fragment (32x16 f16). lanes 0-15: N=lane, K = {0..7, 16..23}; lanes 16-31: N=lane-16,
// K = {8..15, 24..31}. tile is an LDS [N][LDK] (transposed) subtile.
__device__ inline v16h load_b_frag(const f16* tile, int lane) {
    const f16* p = tile + (lane & 15) * LDK + ((lane >> 4) << 3);
    v16h b;
#pragma unroll
    for (int i = 0; i < 8; i++) b[i] = p[i];
#pragma unroll
    for (int i = 0; i < 8; i++) b[8 + i] = p[16 + i];
    return b;
}

__device__ inline float silu_fast(float g) {
    return g * __fdividef(1.0f, 1.0f + __expf(-g));
}

// ---------------- 1) Router: top-1 + sigmoid + per-expert token lists ----------------
__global__ __launch_bounds__(256) void router_kernel(
    const float* __restrict__ hidden, const float* __restrict__ rw,
    int* __restrict__ counts, int* __restrict__ lists, float* __restrict__ score)
{
    const int lane = threadIdx.x & 31;
    const int w = threadIdx.x >> 5;
    const int tok0 = blockIdx.x * 128 + w * 16;
#pragma unroll 1
    for (int tt = 0; tt < 16; tt++) {
        const int t = tok0 + tt;
        const float* hrow = hidden + (size_t)t * Hh;
        float acc[Ee];
#pragma unroll
        for (int e = 0; e < Ee; e++) acc[e] = 0.0f;
#pragma unroll 1
        for (int k = lane; k < Hh; k += 32) {
            const float hv = hrow[k];
#pragma unroll
            for (int e = 0; e < Ee; e++) acc[e] += hv * rw[e * Hh + k];
        }
#pragma unroll
        for (int e = 0; e < Ee; e++) {
#pragma unroll
            for (int off = 16; off > 0; off >>= 1)
                acc[e] += __shfl_xor(acc[e], off, 32);
        }
        if (lane == 0) {
            int be = 0; float bv = acc[0];
#pragma unroll
            for (int e = 1; e < Ee; e++)
                if (acc[e] > bv) { bv = acc[e]; be = e; }
            score[t] = __fdividef(1.0f, 1.0f + __expf(-bv));
            const int pos = atomicAdd(&counts[be], 1);
            lists[be * Tt + pos] = t;
        }
    }
}

// ---------------- 2) gate_up GEMM + fused SiLU*up -> act (f16) ----------------
// ROUTED: A = score[t]*hidden[t] gathered via list, B = gate_up_proj[e] ([H][2I], K-major)
// !ROUTED: A = hidden, B = shared_gate_up_w ([2I][H], N-major)
template <bool ROUTED>
__global__ __launch_bounds__(256) void gateup_silu_kernel(
    const float* __restrict__ hidden, const float* __restrict__ Bw,
    const int* __restrict__ counts, const int* __restrict__ lists,
    const float* __restrict__ score, f16* __restrict__ act)
{
    __shared__ f16 As[2][BM * LDK];
    __shared__ f16 Bg[2][BN * LDK];
    __shared__ f16 Bu[2][BN * LDK];

    const int e = ROUTED ? blockIdx.z : 0;
    const int m0 = blockIdx.x * BM;
    const int n0 = blockIdx.y * BN;

    int cnt = Tt;
    const int* lst = nullptr;
    if (ROUTED) {
        cnt = counts[e];
        if (m0 >= cnt) return;
        lst = lists + (size_t)e * Tt;
    }
    const float* Bbase = ROUTED ? (Bw + (size_t)e * Hh * (2 * Ii)) : Bw;

    const int tid = threadIdx.x;
    const int lane = tid & 31;
    const int w = tid >> 5;
    const int wm = w >> 1, wn = w & 1;   // 4x2 waves -> 128x64 block tile

    // A-load mapping: 128 rows x 8 float4 per row = 1024 float4 / 256 thr
    const int aq = tid & 7;       // float4 index in row (k = aq*4)
    const int ar0 = tid >> 3;     // rows ar0, +32, +64, +96
    int atok[4]; float ascale[4];
#pragma unroll
    for (int i = 0; i < 4; i++) {
        const int gm = m0 + ar0 + i * 32;
        if (ROUTED) {
            const int idx = (gm < cnt) ? gm : (cnt - 1);
            const int t = lst[idx];
            atok[i] = t;
            ascale[i] = (gm < cnt) ? score[t] : 0.0f;
        } else {
            atok[i] = gm; ascale[i] = 1.0f;
        }
    }

    // register staging for ping-pong pipeline
    float4 aReg[4], bgReg[2], buReg[2];

    auto load_regs = [&](int k0) {
#pragma unroll
        for (int i = 0; i < 4; i++)
            aReg[i] = *(const float4*)(hidden + (size_t)atok[i] * Hh + k0 + aq * 4);
        if (ROUTED) {
#pragma unroll
            for (int kk = 0; kk < 2; kk++) {
                const int k = (tid >> 4) + kk * 16;
                const int n = (tid & 15) * 4;
                const float* rowp = Bbase + (size_t)(k0 + k) * (2 * Ii) + n0 + n;
                bgReg[kk] = *(const float4*)(rowp);
                buReg[kk] = *(const float4*)(rowp + Ii);
            }
        } else {
#pragma unroll
            for (int nn = 0; nn < 2; nn++) {
                const int n = (tid >> 3) + nn * 32;
                const int k = (tid & 7) * 4;
                bgReg[nn] = *(const float4*)(Bw + (size_t)(n0 + n) * Hh + k0 + k);
                buReg[nn] = *(const float4*)(Bw + (size_t)(Ii + n0 + n) * Hh + k0 + k);
            }
        }
    };

    auto store_tiles = [&](int buf) {
#pragma unroll
        for (int i = 0; i < 4; i++) {
            f16* dst = As[buf] + (ar0 + i * 32) * LDK + aq * 4;
            dst[0] = (f16)(aReg[i].x * ascale[i]); dst[1] = (f16)(aReg[i].y * ascale[i]);
            dst[2] = (f16)(aReg[i].z * ascale[i]); dst[3] = (f16)(aReg[i].w * ascale[i]);
        }
        if (ROUTED) {
#pragma unroll
            for (int kk = 0; kk < 2; kk++) {
                const int k = (tid >> 4) + kk * 16;
                const int n = (tid & 15) * 4;
                Bg[buf][(n + 0) * LDK + k] = (f16)bgReg[kk].x;
                Bg[buf][(n + 1) * LDK + k] = (f16)bgReg[kk].y;
                Bg[buf][(n + 2) * LDK + k] = (f16)bgReg[kk].z;
                Bg[buf][(n + 3) * LDK + k] = (f16)bgReg[kk].w;
                Bu[buf][(n + 0) * LDK + k] = (f16)buReg[kk].x;
                Bu[buf][(n + 1) * LDK + k] = (f16)buReg[kk].y;
                Bu[buf][(n + 2) * LDK + k] = (f16)buReg[kk].z;
                Bu[buf][(n + 3) * LDK + k] = (f16)buReg[kk].w;
            }
        } else {
#pragma unroll
            for (int nn = 0; nn < 2; nn++) {
                const int n = (tid >> 3) + nn * 32;
                const int k = (tid & 7) * 4;
                f16* dg = Bg[buf] + n * LDK + k;
                dg[0] = (f16)bgReg[nn].x; dg[1] = (f16)bgReg[nn].y;
                dg[2] = (f16)bgReg[nn].z; dg[3] = (f16)bgReg[nn].w;
                f16* du = Bu[buf] + n * LDK + k;
                du[0] = (f16)buReg[nn].x; du[1] = (f16)buReg[nn].y;
                du[2] = (f16)buReg[nn].z; du[3] = (f16)buReg[nn].w;
            }
        }
    };

    v8f accg[2][2], accu[2][2];
    const v8f vzero = {0.f, 0.f, 0.f, 0.f, 0.f, 0.f, 0.f, 0.f};
#pragma unroll
    for (int mi = 0; mi < 2; mi++)
#pragma unroll
        for (int ni = 0; ni < 2; ni++) { accg[mi][ni] = vzero; accu[mi][ni] = vzero; }

    // prologue: tile 0 -> buffer 0
    load_regs(0);
    store_tiles(0);
    __syncthreads();

    int buf = 0;
#pragma unroll 1
    for (int k0 = 0; k0 < Hh; k0 += BK) {
        const bool has_next = (k0 + BK) < Hh;
        if (has_next) {
            load_regs(k0 + BK);                          // issue global loads
            if (k0 + 2 * BK < Hh)                        // GL2 prefetch of tile-after-next
                __builtin_prefetch(Bbase + (size_t)(ROUTED ? (k0 + 2 * BK) * (2 * Ii) + n0
                                                           : (size_t)n0 * Hh + k0 + 2 * BK), 0, 1);
        }
        // compute current tile (overlaps global-load latency)
        v16h bgf[2], buf2[2];
#pragma unroll
        for (int ni = 0; ni < 2; ni++) {
            bgf[ni]  = load_b_frag(Bg[buf] + (wn * 32 + ni * 16) * LDK, lane);
            buf2[ni] = load_b_frag(Bu[buf] + (wn * 32 + ni * 16) * LDK, lane);
        }
#pragma unroll
        for (int mi = 0; mi < 2; mi++) {
            const v16h a = load_a_frag(As[buf] + (wm * 32 + mi * 16) * LDK, lane);
#pragma unroll
            for (int ni = 0; ni < 2; ni++) {
                accg[mi][ni] = wmma32(a, bgf[ni], accg[mi][ni]);
                accu[mi][ni] = wmma32(a, buf2[ni], accu[mi][ni]);
            }
        }
        if (has_next) store_tiles(buf ^ 1);              // commit staged regs
        __syncthreads();
        buf ^= 1;
    }

    // ---- epilogue: act = silu(gate)*up, f16, token-indexed rows ----
#pragma unroll
    for (int mi = 0; mi < 2; mi++)
#pragma unroll
        for (int ni = 0; ni < 2; ni++) {
            const int col = n0 + wn * 32 + ni * 16 + (lane & 15);
            const int rbase = wm * 32 + mi * 16 + ((lane >> 4) << 3);
#pragma unroll
            for (int v = 0; v < 8; v++) {
                const int gm = m0 + rbase + v;
                if (ROUTED && gm >= cnt) continue;
                const float a = silu_fast(accg[mi][ni][v]) * accu[mi][ni][v];
                const int row = ROUTED ? lst[gm] : gm;
                act[(size_t)row * Ii + col] = (f16)a;
            }
        }
}

// ---------------- 3) down GEMM: C = act(f16) x down, write or accumulate ----------------
// ROUTED: B = down_proj[e] ([I][H], K-major), out[tok] += C   (runs after shared pass)
// !ROUTED: B = shared_down_w ([H][I], N-major), out[t] = C
template <bool ROUTED>
__global__ __launch_bounds__(256) void down_kernel(
    const f16* __restrict__ act, const float* __restrict__ Bw,
    const int* __restrict__ counts, const int* __restrict__ lists,
    float* __restrict__ out)
{
    __shared__ f16 As[2][BM * LDK];
    __shared__ f16 Bt[2][BN * LDK];

    const int e = ROUTED ? blockIdx.z : 0;
    const int m0 = blockIdx.x * BM;
    const int n0 = blockIdx.y * BN;

    int cnt = Tt;
    const int* lst = nullptr;
    if (ROUTED) {
        cnt = counts[e];
        if (m0 >= cnt) return;
        lst = lists + (size_t)e * Tt;
    }
    const float* Bbase = ROUTED ? (Bw + (size_t)e * Ii * Hh) : Bw;

    const int tid = threadIdx.x;
    const int lane = tid & 31;
    const int w = tid >> 5;
    const int wm = w >> 1, wn = w & 1;

    // A-load: 128 rows x 4 chunks(8 halves) = 512 chunks / 256 thr -> 2 each
    const int aq = tid & 3;
    const int ar0 = tid >> 2;   // rows ar0, ar0+64
    int atok[2];
#pragma unroll
    for (int i = 0; i < 2; i++) {
        const int gm = m0 + ar0 + i * 64;
        if (ROUTED) {
            const int idx = (gm < cnt) ? gm : (cnt - 1);
            atok[i] = lst[idx];
        } else atok[i] = gm;
    }

    float4 aReg[2], bReg[2];

    auto load_regs = [&](int k0) {
#pragma unroll
        for (int i = 0; i < 2; i++)
            aReg[i] = *(const float4*)(act + (size_t)atok[i] * Ii + k0 + aq * 8);
        if (ROUTED) {
#pragma unroll
            for (int kk = 0; kk < 2; kk++) {
                const int k = (tid >> 4) + kk * 16;
                const int n = (tid & 15) * 4;
                bReg[kk] = *(const float4*)(Bbase + (size_t)(k0 + k) * Hh + n0 + n);
            }
        } else {
#pragma unroll
            for (int nn = 0; nn < 2; nn++) {
                const int n = (tid >> 3) + nn * 32;
                const int k = (tid & 7) * 4;
                bReg[nn] = *(const float4*)(Bw + (size_t)(n0 + n) * Ii + k0 + k);
            }
        }
    };

    auto store_tiles = [&](int buf) {
#pragma unroll
        for (int i = 0; i < 2; i++) {
            const f16* h = (const f16*)&aReg[i];
            f16* dst = As[buf] + (ar0 + i * 64) * LDK + aq * 8;
#pragma unroll
            for (int j = 0; j < 8; j++) dst[j] = h[j];
        }
        if (ROUTED) {
#pragma unroll
            for (int kk = 0; kk < 2; kk++) {
                const int k = (tid >> 4) + kk * 16;
                const int n = (tid & 15) * 4;
                Bt[buf][(n + 0) * LDK + k] = (f16)bReg[kk].x;
                Bt[buf][(n + 1) * LDK + k] = (f16)bReg[kk].y;
                Bt[buf][(n + 2) * LDK + k] = (f16)bReg[kk].z;
                Bt[buf][(n + 3) * LDK + k] = (f16)bReg[kk].w;
            }
        } else {
#pragma unroll
            for (int nn = 0; nn < 2; nn++) {
                const int n = (tid >> 3) + nn * 32;
                const int k = (tid & 7) * 4;
                f16* d = Bt[buf] + n * LDK + k;
                d[0] = (f16)bReg[nn].x; d[1] = (f16)bReg[nn].y;
                d[2] = (f16)bReg[nn].z; d[3] = (f16)bReg[nn].w;
            }
        }
    };

    v8f acc[2][2];
    const v8f vzero = {0.f, 0.f, 0.f, 0.f, 0.f, 0.f, 0.f, 0.f};
#pragma unroll
    for (int mi = 0; mi < 2; mi++)
#pragma unroll
        for (int ni = 0; ni < 2; ni++) acc[mi][ni] = vzero;

    load_regs(0);
    store_tiles(0);
    __syncthreads();

    int buf = 0;
#pragma unroll 1
    for (int k0 = 0; k0 < Ii; k0 += BK) {
        const bool has_next = (k0 + BK) < Ii;
        if (has_next) {
            load_regs(k0 + BK);
            if (k0 + 2 * BK < Ii)
                __builtin_prefetch(Bbase + (size_t)(ROUTED ? (k0 + 2 * BK) * Hh + n0
                                                           : (size_t)n0 * Ii + k0 + 2 * BK), 0, 1);
        }
        v16h bf[2];
#pragma unroll
        for (int ni = 0; ni < 2; ni++)
            bf[ni] = load_b_frag(Bt[buf] + (wn * 32 + ni * 16) * LDK, lane);
#pragma unroll
        for (int mi = 0; mi < 2; mi++) {
            const v16h a = load_a_frag(As[buf] + (wm * 32 + mi * 16) * LDK, lane);
#pragma unroll
            for (int ni = 0; ni < 2; ni++)
                acc[mi][ni] = wmma32(a, bf[ni], acc[mi][ni]);
        }
        if (has_next) store_tiles(buf ^ 1);
        __syncthreads();
        buf ^= 1;
    }

#pragma unroll
    for (int mi = 0; mi < 2; mi++)
#pragma unroll
        for (int ni = 0; ni < 2; ni++) {
            const int col = n0 + wn * 32 + ni * 16 + (lane & 15);
            const int rbase = wm * 32 + mi * 16 + ((lane >> 4) << 3);
#pragma unroll
            for (int v = 0; v < 8; v++) {
                const int gm = m0 + rbase + v;
                if (ROUTED && gm >= cnt) continue;
                const float c = acc[mi][ni][v];
                if (ROUTED) {
                    const int row = lst[gm];
                    out[(size_t)row * Hh + col] += c;  // unique writer (K=1)
                } else {
                    out[(size_t)gm * Hh + col] = c;
                }
            }
        }
}

// ---------------- launch ----------------
extern "C" void kernel_launch(void* const* d_in, const int* in_sizes, int n_in,
                              void* d_out, int out_size, void* d_ws, size_t ws_size,
                              hipStream_t stream) {
    const float* hidden   = (const float*)d_in[0];   // [T,H]
    const float* router_w = (const float*)d_in[1];   // [E,H]
    const float* gate_up  = (const float*)d_in[2];   // [E,H,2I]
    const float* down     = (const float*)d_in[3];   // [E,I,H]
    const float* sh_gu    = (const float*)d_in[4];   // [2I,H]
    const float* sh_dn    = (const float*)d_in[5];   // [H,I]
    float* out = (float*)d_out;                      // [T,H] f32
    char* ws = (char*)d_ws;

    int*   counts = (int*)ws;                                   // 16 ints
    int*   lists  = (int*)(ws + 4096);                          // E*T ints = 256KB
    float* score  = (float*)(ws + 4096 + (size_t)Ee * Tt * 4);  // T f32 = 16KB
    f16*   actR   = (f16*)(ws + 282624);                        // T*I f16 = 8MB
    f16*   actS   = actR + (size_t)Tt * Ii;                     // T*I f16 = 8MB

    hipMemsetAsync(counts, 0, Ee * sizeof(int), stream);

    router_kernel<<<Tt / 128, 256, 0, stream>>>(hidden, router_w, counts, lists, score);

    // shared expert: gate_up + silu
    gateup_silu_kernel<false><<<dim3(Tt / BM, Ii / BN), 256, 0, stream>>>(
        hidden, sh_gu, nullptr, nullptr, nullptr, actS);
    // routed experts: gate_up + silu (A pre-scaled by sigmoid score)
    gateup_silu_kernel<true><<<dim3(Tt / BM, Ii / BN, Ee), 256, 0, stream>>>(
        hidden, gate_up, counts, lists, score, actR);

    // shared down: out = shared
    down_kernel<false><<<dim3(Tt / BM, Hh / BN), 256, 0, stream>>>(
        actS, sh_dn, nullptr, nullptr, out);
    // routed down: out += routed (stream-ordered after shared write)
    down_kernel<true><<<dim3(Tt / BM, Hh / BN, Ee), 256, 0, stream>>>(
        actR, down, counts, lists, out);
}
